// LPN_56684978373020
// MI455X (gfx1250) — matrix operand order repaired and checked by reading, not compiled, not run on hardware
//
#include <hip/hip_runtime.h>
#include <math.h>

// ---------------------------------------------------------------------------
// Types for CDNA5 WMMA
// ---------------------------------------------------------------------------
typedef __attribute__((ext_vector_type(16))) __bf16 v16bf;
typedef __attribute__((ext_vector_type(8)))  float  v8f;
typedef __attribute__((ext_vector_type(4)))  int    v4i;

union FragB16 {
    uint4 u4[2];
    v16bf v;
};
union Pack8 {
    __bf16 h[8];
    uint4  u;
};

// Problem constants
#define NROWS 10000
#define LDK   10016   // 10000 padded up to a multiple of 32
#define HID   256
#define ODIM  128

// ---------------------------------------------------------------------------
// CDNA5 async global->LDS copy (ASYNCcnt) with synchronous fallback
// ---------------------------------------------------------------------------
#if __has_builtin(__builtin_amdgcn_global_load_async_to_lds_b128)
#define USE_ASYNC_LDS 1
typedef __attribute__((address_space(3))) v4i lds_v4i;
#else
#define USE_ASYNC_LDS 0
#endif

__device__ __forceinline__ void copy_b128_to_lds(const __bf16* gsrc, __bf16* ldst) {
#if USE_ASYNC_LDS
    __builtin_amdgcn_global_load_async_to_lds_b128((v4i*)gsrc, (lds_v4i*)ldst, 0, 0);
#else
    *(uint4*)ldst = *(const uint4*)gsrc;
#endif
}

__device__ __forceinline__ void async_copies_wait() {
#if USE_ASYNC_LDS
#if __has_builtin(__builtin_amdgcn_s_wait_asynccnt)
    __builtin_amdgcn_s_wait_asynccnt(0);
#else
    asm volatile("s_wait_asynccnt 0" ::: "memory");
#endif
#endif
}

// ---------------------------------------------------------------------------
// f32 -> bf16 conversion of g into K-padded layout [NROWS x LDK]
// Vectorized: 8 elements (two float4 loads -> one b128 store) per thread.
// ---------------------------------------------------------------------------
__global__ void cvt_g_kernel(const float* __restrict__ g, __bf16* __restrict__ out) {
    size_t chunk = (size_t)blockIdx.x * blockDim.x + threadIdx.x;
    size_t total = (size_t)NROWS * (LDK / 8);
    if (chunk >= total) return;
    int r  = (int)(chunk / (LDK / 8));
    int c8 = (int)(chunk % (LDK / 8)) * 8;
    Pack8 p;
    if (c8 + 8 <= NROWS) {     // NROWS % 8 == 0, so this covers every real column
        const float4* src = (const float4*)(g + (size_t)r * NROWS + c8);
        float4 a = src[0], b = src[1];
        p.h[0] = (__bf16)a.x; p.h[1] = (__bf16)a.y; p.h[2] = (__bf16)a.z; p.h[3] = (__bf16)a.w;
        p.h[4] = (__bf16)b.x; p.h[5] = (__bf16)b.y; p.h[6] = (__bf16)b.z; p.h[7] = (__bf16)b.w;
    } else {                   // K padding
#pragma unroll
        for (int i = 0; i < 8; ++i) p.h[i] = (__bf16)0.0f;
    }
    *(uint4*)(out + chunk * 8) = p.u;
}

// f32 activation [NROWS x D] -> bf16 row-padded [LDK x D] (pad rows zeroed)
template <int D>
__global__ void cvt_x_kernel(const float* __restrict__ x, __bf16* __restrict__ out) {
    size_t chunk = (size_t)blockIdx.x * blockDim.x + threadIdx.x;
    size_t total = (size_t)LDK * (D / 8);
    if (chunk >= total) return;
    int r = (int)(chunk / (D / 8));
    Pack8 p;
    if (r < NROWS) {
        const float4* src = (const float4*)(x + chunk * 8);
        float4 a = src[0], b = src[1];
        p.h[0] = (__bf16)a.x; p.h[1] = (__bf16)a.y; p.h[2] = (__bf16)a.z; p.h[3] = (__bf16)a.w;
        p.h[4] = (__bf16)b.x; p.h[5] = (__bf16)b.y; p.h[6] = (__bf16)b.z; p.h[7] = (__bf16)b.w;
    } else {
#pragma unroll
        for (int i = 0; i < 8; ++i) p.h[i] = (__bf16)0.0f;
    }
    *(uint4*)(out + chunk * 8) = p.u;
}

// ---------------------------------------------------------------------------
// Big GEMM:  Y[NROWS x D] (f32) = g_bf16[NROWS x LDK] @ X_bf16[LDK x D]
// 256 threads = 8 waves; wave w covers row strip (w>>1)*16 of a 64-row block
// tile and column half (w&1)*(D/2).  Double-buffered LDS tiles filled by
// async global->LDS copies (ASYNCcnt); B fragments loaded in groups of 4
// distinct register sets so one s_wait_dscnt covers 4 WMMAs.
// EPI: fused APPNP step  Y = 0.5 * (g@X) + 0.5 * H
// NOTE: A-tile loads are unguarded on the last block: the OOB rows read
// adjacent workspace (mapped memory) and the garbage lands only in C rows
// >= NROWS, whose stores are guarded.
// ---------------------------------------------------------------------------
template <int D, bool EPI>
__global__ __launch_bounds__(256)
void gemm_big_kernel(const __bf16* __restrict__ gbf,
                     const __bf16* __restrict__ xbf,
                     const float*  __restrict__ H,
                     float* __restrict__ Y) {
    constexpr int NTW   = D / 32;   // 16-wide N tiles per wave (half of D)
    constexpr int ASTR  = 40;       // A LDS row stride (bf16): 80B rows, 16B aligned
    constexpr int BSTR  = D + 8;    // B LDS row stride (bf16): de-aligns banks
    constexpr int NSTEP = LDK / 32;
    constexpr int BCH   = (32 * (D / 8)) / 256;  // B b128 chunks per thread (exact)

    __shared__ __align__(16) __bf16 As[2][64 * ASTR];
    __shared__ __align__(16) __bf16 Bs[2][32 * BSTR];

    const int tid    = threadIdx.x;       // 0..255
    const int wave   = tid >> 5;          // 0..7
    const int lane   = tid & 31;          // 0..31
    const int mstrip = wave >> 1;         // 0..3 : 16-row strip in block tile
    const int nbase  = (wave & 1) * (D / 2);
    const int m0     = blockIdx.x * 64;   // block row base

    v8f acc[NTW];
#pragma unroll
    for (int t = 0; t < NTW; ++t)
        acc[t] = v8f{0.f, 0.f, 0.f, 0.f, 0.f, 0.f, 0.f, 0.f};

    auto issue_tile = [&](int k0, int buf) {
        // A tile: 64x32 bf16 = 256 b128 chunks, exactly 1 per thread
        {
            int r  = tid >> 2;
            int cc = (tid & 3) * 8;
            copy_b128_to_lds(gbf + (size_t)(m0 + r) * LDK + k0 + cc,
                             &As[buf][r * ASTR + cc]);
        }
        // B tile: 32xD bf16, BCH chunks per thread (row-padded source)
#pragma unroll
        for (int u = 0; u < BCH; ++u) {
            int i  = tid + u * 256;
            int r  = i / (D / 8);
            int cc = (i % (D / 8)) * 8;
            copy_b128_to_lds(xbf + (size_t)(k0 + r) * D + cc,
                             &Bs[buf][r * BSTR + cc]);
        }
    };

    issue_tile(0, 0);

    const int mrow  = mstrip * 16 + (lane & 15);
    const int khalf = lane >> 4;

    for (int ks = 0; ks < NSTEP; ++ks) {
        const int buf = ks & 1;
        async_copies_wait();     // this wave's copies into buf have landed
        __syncthreads();         // everyone's copies landed; prior reads done
        if (ks + 1 < NSTEP) issue_tile((ks + 1) * 32, buf ^ 1);

        // A fragment (16x32 bf16, ISA layout):
        //  lane<16 : M=lane,    K chunks {0..7},{16..23}
        //  lane>=16: M=lane-16, K chunks {8..15},{24..31}
        const __bf16* arow = &As[buf][mrow * ASTR];
        FragB16 a;
        a.u4[0] = *(const uint4*)(arow + khalf * 8);
        a.u4[1] = *(const uint4*)(arow + 16 + khalf * 8);

        // B fragments (32x16: lane l holds column N=l%16, K in same order as A).
        // Load 4 fragments into distinct registers, then 4 WMMAs: one ds
        // drain per 4 matrix ops.
        const __bf16* brow = &Bs[buf][nbase + (lane & 15)];
#pragma unroll
        for (int tg = 0; tg < NTW; tg += 4) {
            FragB16 b[4];
#pragma unroll
            for (int u = 0; u < 4; ++u) {
                const __bf16* bp = brow + (tg + u) * 16;
#pragma unroll
                for (int s = 0; s < 16; ++s) {
                    int k = (s < 8) ? (khalf * 8 + s) : (16 + khalf * 8 + (s - 8));
                    b[u].v[s] = bp[k * BSTR];
                }
            }
#pragma unroll
            for (int u = 0; u < 4; ++u) {
                acc[tg + u] = __builtin_amdgcn_wmma_f32_16x16x32_bf16(
                    false, a.v, false, b[u].v, (short)0, acc[tg + u], false, false);
            }
        }
    }

    // C layout: VGPR r, lanes0-15: M=r, N=lane; lanes16-31: M=r+8, N=lane-16
    const int nlo = lane & 15;
    const int mhi = (lane >> 4) * 8;
#pragma unroll
    for (int t = 0; t < NTW; ++t) {
#pragma unroll
        for (int r = 0; r < 8; ++r) {
            int row = m0 + mstrip * 16 + r + mhi;
            int col = nbase + t * 16 + nlo;
            if (row < NROWS) {
                float v = acc[t][r];
                size_t idx = (size_t)row * D + col;
                if (EPI) v = 0.5f * v + 0.5f * H[idx];
                Y[idx] = v;
            }
        }
    }
}

// ---------------------------------------------------------------------------
// Small GEMM + activation:  Out[N x DOUT] = act(Yin[N x DIN] @ W^T)
// One block per row; the activation row is staged in LDS once.
// W is torch-layout [DOUT x DIN]. ACT: 0 = relu, 1 = sigmoid.
// ---------------------------------------------------------------------------
template <int DIN, int DOUT, int ACT>
__global__ __launch_bounds__(DOUT)
void small_gemm_kernel(const float* __restrict__ Yin,
                       const float* __restrict__ W,
                       float* __restrict__ Out) {
    __shared__ float yrow[DIN];
    const int row = blockIdx.x;
    for (int k = threadIdx.x; k < DIN; k += DOUT)
        yrow[k] = Yin[(size_t)row * DIN + k];
    __syncthreads();
    const int j = threadIdx.x;
    const float* w = W + (size_t)j * DIN;
    float s = 0.0f;
#pragma unroll 8
    for (int k = 0; k < DIN; ++k) s = fmaf(yrow[k], w[k], s);
    Out[(size_t)row * DOUT + j] =
        (ACT == 0) ? fmaxf(s, 0.0f) : 1.0f / (1.0f + __expf(-s));
}

// ---------------------------------------------------------------------------
// Row L2 normalize (HID == 256, one block per row)
// ---------------------------------------------------------------------------
__global__ __launch_bounds__(256)
void rownorm_kernel(const float* __restrict__ X, float* __restrict__ Out) {
    __shared__ float red[256];
    const int row = blockIdx.x;
    const int tid = threadIdx.x;
    float v = X[(size_t)row * HID + tid];
    red[tid] = v * v;
    __syncthreads();
#pragma unroll
    for (int s = 128; s > 0; s >>= 1) {
        if (tid < s) red[tid] += red[tid + s];
        __syncthreads();
    }
    float norm = fmaxf(sqrtf(red[0]), 1e-12f);
    Out[(size_t)row * HID + tid] = v / norm;
}

// ---------------------------------------------------------------------------
// Orchestration
// ---------------------------------------------------------------------------
extern "C" void kernel_launch(void* const* d_in, const int* in_sizes, int n_in,
                              void* d_out, int out_size, void* d_ws, size_t ws_size,
                              hipStream_t stream) {
    const float* g   = (const float*)d_in[0];
    const float* z   = (const float*)d_in[1];
    const float* W1  = (const float*)d_in[2];
    const float* W1_ = (const float*)d_in[3];
    const float* W2  = (const float*)d_in[4];

    float* out_res = (float*)d_out;                        // [10000 x 128]
    float* out_Z   = (float*)d_out + (size_t)NROWS * ODIM; // [10000 x 256]

    // ---- workspace carve-up ----
    char* ws = (char*)d_ws;
    size_t off = 0;
    __bf16* gbf = (__bf16*)(ws + off); off += (size_t)NROWS * LDK * sizeof(__bf16);
    __bf16* xbf = (__bf16*)(ws + off); off += (size_t)LDK * HID * sizeof(__bf16);
    float* f1   = (float*)(ws + off);  off += (size_t)NROWS * HID * sizeof(float);
    float* f2   = (float*)(ws + off);  off += (size_t)NROWS * HID * sizeof(float);
    float* f3   = (float*)(ws + off);  off += (size_t)NROWS * HID * sizeof(float);
    float* z1b  = (float*)(ws + off);  off += (size_t)NROWS * HID * sizeof(float);
    float* zhid = (float*)(ws + off);  off += (size_t)NROWS * HID * sizeof(float);
    float* zn   = (float*)(ws + off);  off += (size_t)NROWS * HID * sizeof(float);
    (void)ws_size;

    const int TB = 256;
    dim3 gemm_grid((NROWS + 63) / 64);
    dim3 gemm_block(256);

    // 1. g -> bf16, K-padded (one-time, amortized across 6 GEMM passes)
    {
        size_t total = (size_t)NROWS * (LDK / 8);
        cvt_g_kernel<<<dim3((unsigned)((total + TB - 1) / TB)), TB, 0, stream>>>(g, gbf);
    }

    // 2. z1 = relu((g @ z) @ W1^T)
    {
        size_t total = (size_t)LDK * (ODIM / 8);
        cvt_x_kernel<ODIM><<<dim3((unsigned)((total + TB - 1) / TB)), TB, 0, stream>>>(z, xbf);
        gemm_big_kernel<ODIM, false><<<gemm_grid, gemm_block, 0, stream>>>(gbf, xbf, nullptr, f1);
        small_gemm_kernel<ODIM, HID, 0><<<dim3(NROWS), HID, 0, stream>>>(f1, W1, z1b);
    }

    // 3. z_ = relu((g @ z1) @ W1_^T)
    {
        size_t total = (size_t)LDK * (HID / 8);
        cvt_x_kernel<HID><<<dim3((unsigned)((total + TB - 1) / TB)), TB, 0, stream>>>(z1b, xbf);
        gemm_big_kernel<HID, false><<<gemm_grid, gemm_block, 0, stream>>>(gbf, xbf, nullptr, f2);
        small_gemm_kernel<HID, HID, 0><<<dim3(NROWS), HID, 0, stream>>>(f2, W1_, zhid);
    }

    // 4. zn = row-normalize(z_)
    rownorm_kernel<<<dim3(NROWS), 256, 0, stream>>>(zhid, zn);

    // 5. APPNP: Z = 0.5*(g@Z) + 0.5*zn, three iterations (last writes d_out)
    {
        size_t total = (size_t)LDK * (HID / 8);
        cvt_x_kernel<HID><<<dim3((unsigned)((total + TB - 1) / TB)), TB, 0, stream>>>(zn, xbf);
        gemm_big_kernel<HID, true><<<gemm_grid, gemm_block, 0, stream>>>(gbf, xbf, zn, f2);

        cvt_x_kernel<HID><<<dim3((unsigned)((total + TB - 1) / TB)), TB, 0, stream>>>(f2, xbf);
        gemm_big_kernel<HID, true><<<gemm_grid, gemm_block, 0, stream>>>(gbf, xbf, zn, f3);

        cvt_x_kernel<HID><<<dim3((unsigned)((total + TB - 1) / TB)), TB, 0, stream>>>(f3, xbf);
        gemm_big_kernel<HID, true><<<gemm_grid, gemm_block, 0, stream>>>(gbf, xbf, zn, out_Z);
    }

    // 6. res = sigmoid((g @ z_) @ W2^T)
    {
        size_t total = (size_t)LDK * (HID / 8);
        cvt_x_kernel<HID><<<dim3((unsigned)((total + TB - 1) / TB)), TB, 0, stream>>>(zhid, xbf);
        gemm_big_kernel<HID, false><<<gemm_grid, gemm_block, 0, stream>>>(gbf, xbf, nullptr, f2);
        small_gemm_kernel<HID, ODIM, 1><<<dim3(NROWS), ODIM, 0, stream>>>(f2, W2, out_res);
    }
}
